// AlchemicalGAT_30382598652273
// MI455X (gfx1250) — matrix operand order repaired and checked by reading, not compiled, not run on hardware
//
#include <hip/hip_runtime.h>
#include <hip/hip_bf16.h>

// ---------------------------------------------------------------------------
// Types for CDNA5 WMMA
// ---------------------------------------------------------------------------
typedef __attribute__((ext_vector_type(16))) __bf16 v16bf;
typedef __attribute__((ext_vector_type(8)))  float  v8f;

union FragAB { uint4 q[2]; v16bf v; };   // 32 bytes: 16 bf16 per lane
union AccU   { v8f v; float f[8]; };     // 32 bytes: 8 f32 per lane

__device__ __forceinline__ unsigned short f2bf(float f) {
    union { float f; unsigned u; } v; v.f = f;
    unsigned r = v.u + 0x7FFFu + ((v.u >> 16) & 1u);   // round-to-nearest-even
    return (unsigned short)(r >> 16);
}
__device__ __forceinline__ float bf2f(unsigned short h) {
    union { unsigned u; float f; } v; v.u = ((unsigned)h) << 16;
    return v.f;
}
__device__ __forceinline__ float silu(float x) {
    return x / (1.0f + __expf(-x));
}
// order-preserving float <-> uint map for atomic segment-max
__device__ __forceinline__ unsigned encf(float f) {
    unsigned u = __float_as_uint(f);
    return (u & 0x80000000u) ? ~u : (u | 0x80000000u);
}
__device__ __forceinline__ float decf(unsigned u) {
    return __uint_as_float((u & 0x80000000u) ? (u & 0x7FFFFFFFu) : ~u);
}

#define CCH 4   // channels

// ---------------------------------------------------------------------------
// 1) LayerNorm over F=128, emit bf16 rows [N*C, 128]. One wave32 per row.
// ---------------------------------------------------------------------------
__global__ __launch_bounds__(256)
void ln_kernel(const float* __restrict__ x, const float* __restrict__ gamma,
               const float* __restrict__ beta, unsigned short* __restrict__ out,
               int rows)
{
    int r    = blockIdx.x * 8 + (threadIdx.x >> 5);
    int lane = threadIdx.x & 31;
    if (r >= rows) return;
    const float4 v = *(const float4*)(x + (size_t)r * 128 + lane * 4);
    float s = v.x + v.y + v.z + v.w;
    for (int o = 16; o > 0; o >>= 1) s += __shfl_xor(s, o, 32);
    float mu = s * (1.0f / 128.0f);
    float dx = v.x - mu, dy = v.y - mu, dz = v.z - mu, dw = v.w - mu;
    float q = dx * dx + dy * dy + dz * dz + dw * dw;
    for (int o = 16; o > 0; o >>= 1) q += __shfl_xor(q, o, 32);
    float inv = rsqrtf(q * (1.0f / 128.0f) + 1e-5f);
    const float4 g = *(const float4*)(gamma + lane * 4);
    const float4 b = *(const float4*)(beta  + lane * 4);
    unsigned h0 = f2bf(dx * inv * g.x + b.x);
    unsigned h1 = f2bf(dy * inv * g.y + b.y);
    unsigned h2 = f2bf(dz * inv * g.z + b.z);
    unsigned h3 = f2bf(dw * inv * g.w + b.w);
    uint2 p; p.x = h0 | (h1 << 16); p.y = h2 | (h3 << 16);
    *(uint2*)(out + (size_t)r * 128 + lane * 4) = p;
}

// ---------------------------------------------------------------------------
// 2) Per-channel GEMM via V_WMMA_F32_16X16X32_BF16.
//    A: bf16 [M, C, K] (row per atom/channel), W: f32 [C, K, NOUT].
//    grid = (ceil(M/128), C), block = 256 (8 waves, 16 rows each).
//    OUTMODE 0: f32 raw   OUTMODE 1: SiLU -> bf16
// ---------------------------------------------------------------------------
template<int K, int NOUT, int OUTMODE>
__global__ __launch_bounds__(256)
void gemm_mc(const unsigned short* __restrict__ A, const float* __restrict__ W,
             float* __restrict__ outF, unsigned short* __restrict__ outB, int M)
{
    constexpr int KSTEPS = K / 32;
    constexpr int NT     = NOUT / 16;
    __shared__ uint4 ldsb[(K * NOUT) / 8];   // bf16 B-fragments, pre-swizzled

    const int tid = threadIdx.x;
    const int c   = blockIdx.y;

    // Cooperative load of W[c] into LDS, swizzled to the B-fragment order this kernel uses:
    // lds ushort index = ((kk*NT + t)*32 + lane)*16 + j
    //   holds W[kk*32 + (lane<16?0:16) + j][t*16 + (lane&15)]   (contiguous k, same order as A)
    {
        unsigned short* lb = (unsigned short*)ldsb;
        const float* Wc = W + (size_t)c * K * NOUT;
        for (int idx = tid; idx < K * NOUT; idx += 256) {
            int j    = idx & 15;
            int lane = (idx >> 4) & 31;
            int ft   = idx >> 9;
            int t    = ft % NT;
            int kk   = ft / NT;
            int k    = kk * 32 + ((lane < 16) ? 0 : 16) + j;
            int n    = t * 16 + (lane & 15);
            lb[idx]  = f2bf(Wc[(size_t)k * NOUT + n]);
        }
    }
    __syncthreads();

    const int wave = tid >> 5;
    const int lane = tid & 31;
    const int m0   = blockIdx.x * 128 + wave * 16;
    int row = m0 + (lane & 15);
    if (row >= M) row = M - 1;               // clamp loads; stores are guarded
    const size_t abase = ((size_t)row * CCH + c) * K;
    const int kb0 = (lane < 16) ? 0 : 16;    // A contiguous k = 16h + s (matches B)
    const int kb1 = (lane < 16) ? 8 : 24;

    AccU acc[NT];
    #pragma unroll
    for (int t = 0; t < NT; ++t)
        #pragma unroll
        for (int r = 0; r < 8; ++r) acc[t].f[r] = 0.0f;

    #pragma unroll
    for (int kk = 0; kk < KSTEPS; ++kk) {
        FragAB a;
        a.q[0] = *(const uint4*)(A + abase + kk * 32 + kb0);
        a.q[1] = *(const uint4*)(A + abase + kk * 32 + kb1);
        #pragma unroll
        for (int t = 0; t < NT; ++t) {
            FragAB b;
            const uint4* bp = ldsb + ((size_t)(kk * NT + t) * 32 + lane) * 2;
            b.q[0] = bp[0];
            b.q[1] = bp[1];
            acc[t].v = __builtin_amdgcn_wmma_f32_16x16x32_bf16(
                false, a.v, false, b.v, (short)0, acc[t].v, false, false);
        }
    }

    // C/D layout: VGPR r, lane l -> row m0 + r + (l<16?0:8), col l&15
    const int mr  = m0 + ((lane < 16) ? 0 : 8);
    const int col = lane & 15;
    #pragma unroll
    for (int t = 0; t < NT; ++t) {
        #pragma unroll
        for (int r = 0; r < 8; ++r) {
            int m = mr + r;
            if (m < M) {
                size_t o = ((size_t)m * CCH + c) * NOUT + t * 16 + col;
                float v = acc[t].f[r];
                if (OUTMODE == 0) outF[o] = v;
                else              outB[o] = f2bf(silu(v));
            }
        }
    }
}

// ---------------------------------------------------------------------------
// 3) Attention scores: e_src/e_dst = h . a  per (n,c) row of 64
// ---------------------------------------------------------------------------
__global__ __launch_bounds__(256)
void ekernel(const float* __restrict__ h, const float* __restrict__ asrc,
             const float* __restrict__ adst, float* __restrict__ es,
             float* __restrict__ ed, int rows)
{
    int idx = blockIdx.x * 256 + threadIdx.x;
    if (idx >= rows) return;
    int c = idx & (CCH - 1);
    const float* hp = h + (size_t)idx * 64;
    const float* as = asrc + c * 64;
    const float* ad = adst + c * 64;
    float s = 0.f, d = 0.f;
    #pragma unroll 8
    for (int f = 0; f < 64; ++f) { float v = hp[f]; s += v * as[f]; d += v * ad[f]; }
    es[idx] = s; ed[idx] = d;
}

// ---------------------------------------------------------------------------
// 4) Edge pass 1: leaky-relu logits + segment-max via uint atomicMax
// ---------------------------------------------------------------------------
__global__ __launch_bounds__(256)
void edge_logit(const int* __restrict__ src, const int* __restrict__ dst,
                const float* __restrict__ es, const float* __restrict__ ed,
                float* __restrict__ elog, unsigned* __restrict__ mmax, int E)
{
    int e = blockIdx.x * 256 + threadIdx.x;
    if (e >= E) return;
    int s = src[e], d = dst[e];
    #pragma unroll
    for (int c = 0; c < CCH; ++c) {
        float l = es[s * CCH + c] + ed[d * CCH + c];
        l = (l > 0.0f) ? l : 0.2f * l;
        elog[(size_t)e * CCH + c] = l;
        atomicMax(mmax + d * CCH + c, encf(l));
    }
}

// 5) Edge pass 2: exp(logit - max) + segment-sum denominator
__global__ __launch_bounds__(256)
void edge_exp(const int* __restrict__ dst, float* __restrict__ elog,
              const unsigned* __restrict__ mmax, float* __restrict__ den, int E)
{
    int e = blockIdx.x * 256 + threadIdx.x;
    if (e >= E) return;
    int d = dst[e];
    #pragma unroll
    for (int c = 0; c < CCH; ++c) {
        float ex = __expf(elog[(size_t)e * CCH + c] - decf(mmax[d * CCH + c]));
        elog[(size_t)e * CCH + c] = ex;
        atomicAdd(den + d * CCH + c, ex);
    }
}

// 6) Edge pass 3: scatter alpha * h[src] into agg[dst] (f32 atomics, L2 resident)
__global__ __launch_bounds__(256)
void edge_agg(const int* __restrict__ src, const int* __restrict__ dst,
              const float* __restrict__ elog, const float* __restrict__ den,
              const float* __restrict__ h, float* __restrict__ agg, int EC)
{
    int idx = blockIdx.x * 256 + threadIdx.x;
    if (idx >= EC) return;
    int e = idx >> 2, c = idx & 3;
    int s = src[e], d = dst[e];
    float alpha = elog[(size_t)e * CCH + c] / (den[d * CCH + c] + 1e-16f);
    const float* hp = h   + ((size_t)s * CCH + c) * 64;
    float*       ap = agg + ((size_t)d * CCH + c) * 64;
    #pragma unroll 4
    for (int f = 0; f < 64; ++f) atomicAdd(ap + f, alpha * hp[f]);
}

// 7) SiLU + f32 -> bf16 (feeds next GEMM)
__global__ __launch_bounds__(256)
void silu_bf16(const float* __restrict__ in, unsigned short* __restrict__ out,
               unsigned n)
{
    unsigned i = blockIdx.x * 256u + threadIdx.x;
    if (i >= n) return;
    out[i] = f2bf(silu(in[i]));
}

// 8) Final: 32->1 per channel, channel sum, scaled segment-sum into energies
__global__ __launch_bounds__(256)
void out_kernel(const unsigned short* __restrict__ a, const float* __restrict__ Wout,
                const int* __restrict__ batch, float* __restrict__ out, int N)
{
    int n = blockIdx.x * 256 + threadIdx.x;
    if (n >= N) return;
    float feat = 0.f;
    #pragma unroll
    for (int c = 0; c < CCH; ++c) {
        const unsigned short* ap = a + ((size_t)n * CCH + c) * 32;
        const float* wp = Wout + c * 32;
        #pragma unroll 8
        for (int f = 0; f < 32; ++f) feat += bf2f(ap[f]) * wp[f];
    }
    atomicAdd(out + batch[n], feat * 0.025f);   // 1/(sqrt(4)*20)
}

// ---------------------------------------------------------------------------
// Launcher
// ---------------------------------------------------------------------------
extern "C" void kernel_launch(void* const* d_in, const int* in_sizes, int n_in,
                              void* d_out, int out_size, void* d_ws, size_t ws_size,
                              hipStream_t stream)
{
    const float* x     = (const float*)d_in[0];
    const int*   edge  = (const int*)d_in[1];
    const int*   batch = (const int*)d_in[2];
    const float* gamma = (const float*)d_in[3];
    const float* beta  = (const float*)d_in[4];
    const float* Wc1   = (const float*)d_in[5];
    const float* asrc1 = (const float*)d_in[6];
    const float* adst1 = (const float*)d_in[7];
    const float* Wc2   = (const float*)d_in[8];
    const float* asrc2 = (const float*)d_in[9];
    const float* adst2 = (const float*)d_in[10];
    const float* Wn1   = (const float*)d_in[11];
    const float* Wn2   = (const float*)d_in[12];
    const float* Wout  = (const float*)d_in[13];

    const int N = in_sizes[0] / (CCH * 128);
    const int E = in_sizes[1] / 2;
    const int* src = edge;
    const int* dst = edge + E;
    const int NC = N * CCH;

    // workspace carve-up (256B aligned)
    char* ws = (char*)d_ws;
    size_t off = 0;
    auto carve = [&](size_t bytes) {
        size_t o = off; off += (bytes + 255) & ~(size_t)255; return o;
    };
    unsigned short* A_bf = (unsigned short*)(ws + carve((size_t)NC * 128 * 2)); // GEMM input
    float*          H    = (float*)(ws + carve((size_t)NC * 64 * 4));           // GEMM output h
    float*          AGG  = (float*)(ws + carve((size_t)NC * 64 * 4));           // aggregation
    float*          ES   = (float*)(ws + carve((size_t)NC * 4));
    float*          ED   = (float*)(ws + carve((size_t)NC * 4));
    unsigned*       MMAX = (unsigned*)(ws + carve((size_t)NC * 4));
    float*          DEN  = (float*)(ws + carve((size_t)NC * 4));
    float*          ELOG = (float*)(ws + carve((size_t)E * CCH * 4));

    dim3 gGemm((N + 127) / 128, CCH);
    int gRows  = (NC + 255) / 256;
    int gEdge  = (E + 255) / 256;
    int gEdgeC = (E * CCH + 255) / 256;
    int gFeat  = (NC * 64 + 255) / 256;
    int gAtom  = (N + 255) / 256;

    // --- LayerNorm -> bf16 A
    ln_kernel<<<(NC + 7) / 8, 256, 0, stream>>>(x, gamma, beta, A_bf, NC);

    // --- GAT layer 1 (K=128)
    gemm_mc<128, 64, 0><<<gGemm, 256, 0, stream>>>(A_bf, Wc1, H, nullptr, N);
    ekernel<<<gRows, 256, 0, stream>>>(H, asrc1, adst1, ES, ED, NC);
    hipMemsetAsync(MMAX, 0, (size_t)NC * 4, stream);
    hipMemsetAsync(DEN,  0, (size_t)NC * 4, stream);
    hipMemsetAsync(AGG,  0, (size_t)NC * 64 * 4, stream);
    edge_logit<<<gEdge, 256, 0, stream>>>(src, dst, ES, ED, ELOG, MMAX, E);
    edge_exp<<<gEdge, 256, 0, stream>>>(dst, ELOG, MMAX, DEN, E);
    edge_agg<<<gEdgeC, 256, 0, stream>>>(src, dst, ELOG, DEN, H, AGG, E * CCH);
    silu_bf16<<<gFeat, 256, 0, stream>>>(AGG, A_bf, (unsigned)(NC * 64));

    // --- GAT layer 2 (K=64)
    gemm_mc<64, 64, 0><<<gGemm, 256, 0, stream>>>(A_bf, Wc2, H, nullptr, N);
    ekernel<<<gRows, 256, 0, stream>>>(H, asrc2, adst2, ES, ED, NC);
    hipMemsetAsync(MMAX, 0, (size_t)NC * 4, stream);
    hipMemsetAsync(DEN,  0, (size_t)NC * 4, stream);
    hipMemsetAsync(AGG,  0, (size_t)NC * 64 * 4, stream);
    edge_logit<<<gEdge, 256, 0, stream>>>(src, dst, ES, ED, ELOG, MMAX, E);
    edge_exp<<<gEdge, 256, 0, stream>>>(dst, ELOG, MMAX, DEN, E);
    edge_agg<<<gEdgeC, 256, 0, stream>>>(src, dst, ELOG, DEN, H, AGG, E * CCH);
    silu_bf16<<<gFeat, 256, 0, stream>>>(AGG, A_bf, (unsigned)(NC * 64));

    // --- MLP: 64->64 (SiLU, bf16 into H region), 64->32 (SiLU, bf16 into A region)
    gemm_mc<64, 64, 1><<<gGemm, 256, 0, stream>>>(A_bf, Wn1, nullptr,
                                                  (unsigned short*)H, N);
    gemm_mc<64, 32, 1><<<gGemm, 256, 0, stream>>>((const unsigned short*)H, Wn2,
                                                  nullptr, A_bf, N);

    // --- energies
    hipMemsetAsync(d_out, 0, (size_t)out_size * 4, stream);
    out_kernel<<<gAtom, 256, 0, stream>>>(A_bf, Wout, batch, (float*)d_out, N);
}
